// DenseSum_28698971471971
// MI455X (gfx1250) — matrix operand order repaired and checked by reading, not compiled, not run on hardware
//
#include <hip/hip_runtime.h>
#include <hip/hip_bf16.h>

// ---------------------------------------------------------------------------
// out[s,d,b,j] = logsumexp_i( x[s,d,b,i] + log_softmax_i(log(acc))[s,d,i,j] )
//
// Decomposed as:  E[b,i] = exp(x - rowmax),  W[i,j] = acc/colmax  (both in (0,1])
//                 Y = E @ W  (f16 WMMA, f32 accumulate)
//                 out = log(Y) + rowmax[b] + log(colmax[j]) - log(colsum[j])
//
// One workgroup (512 thr = 16 wave32) per (scope,decomp) problem; whole
// 256x256x256 GEMM staged in LDS (E: 132KB, W^T: 132KB  < 320KB/WGP).
// Each wave: 64x64 C tile = 4x4 v_wmma_f32_16x16x32_f16 accumulators.
//
// HBM traffic: x 64MiB + acc 64MiB (read ONCE; normalization done in LDS)
// + out 64MiB = 192MiB  ->  ~8.2us at 23.3 TB/s roof. exp/log use raw
// v_exp_f32/v_log_f32 (operands provably never denormal) to keep VALU lean.
// ---------------------------------------------------------------------------

typedef __attribute__((ext_vector_type(16))) _Float16 v16h;
typedef __attribute__((ext_vector_type(8)))  _Float16 v8h;
typedef __attribute__((ext_vector_type(8)))  float    v8f;

#define N256       256
#define LDSS       264                      // padded f16 row stride (528B, 16B-aligned)
#define XE_ELEMS   (N256 * LDSS)
#define SMEM_BYTES (2 * XE_ELEMS * 2 + (256 * 3 + 512 * 2) * 4)   // 277,504 B

#define LOG2E 1.44269504088896340736f
#define LN2   0.69314718055994530942f

// A fragment (16x32 f16, MxK): lanes 0-15 row M=l hold K[0..7] in v0-3, K[16..23]
// in v4-7; lanes 16-31 row M=l-16 hold K[8..15], K[24..31].
__device__ __forceinline__ v16h load_a_frag(const _Float16* xe, int rowbase, int kb, int lane) {
  int row = rowbase + (lane & 15);
  int k0  = kb + ((lane >> 4) << 3);
  const _Float16* p = xe + row * LDSS + k0;
  v8h lo = *(const v8h*)(p);
  v8h hi = *(const v8h*)(p + 16);
  return __builtin_shufflevector(lo, hi, 0,1,2,3,4,5,6,7,8,9,10,11,12,13,14,15);
}

// B fragment (32x16 f16, KxN): lane n<16 holds column N=n, K[0..7], K[16..23];
// lanes 16-31 hold column N=n-16, K[8..15], K[24..31]. W is stored transposed (K-contiguous
// per column) in LDS so this is two aligned 16B loads.
__device__ __forceinline__ v16h load_b_frag(const _Float16* we, int colbase, int kb, int lane) {
  int col = colbase + (lane & 15);
  int k0  = kb + ((lane >> 4) << 3);
  const _Float16* p = we + col * LDSS + k0;
  v8h lo = *(const v8h*)(p);
  v8h hi = *(const v8h*)(p + 16);
  return __builtin_shufflevector(lo, hi, 0,1,2,3,4,5,6,7,8,9,10,11,12,13,14,15);
}

__global__ __launch_bounds__(512)
void densesum_logmm_wmma(const float* __restrict__ X,
                         const float* __restrict__ A,
                         float* __restrict__ O) {
  extern __shared__ __align__(16) char smem[];
  _Float16* xe     = (_Float16*)smem;            // E[b][i]   256 x 264 f16
  _Float16* we     = xe + XE_ELEMS;              // W^T[j][i] 256 x 264 f16
  float*    xmaxs  = (float*)(we + XE_ELEMS);    // rowmax[b]           256
  float*    cterm  = xmaxs + 256;                // log(cmax)-log(csum) 256
  float*    cmaxs  = cterm + 256;                // colmax[j]           256
  float*    redmax = cmaxs + 256;                // 512
  float*    redsum = redmax + 512;               // 512

  const int    tid  = threadIdx.x;
  const int    lane = tid & 31;
  const int    wv   = tid >> 5;                          // 0..15
  const size_t base = (size_t)blockIdx.x * (N256 * N256);

  // ---- Phase 1a: E[b,i] = exp(x - rowmax) as f16; one wave per 16 rows ----
  for (int r = wv * 16; r < wv * 16 + 16; ++r) {
    const float* xr = X + base + (size_t)r * N256 + lane;
    float v[8];
    float m = -3.402823466e38f;
#pragma unroll
    for (int u = 0; u < 8; ++u) { v[u] = xr[u * 32]; m = fmaxf(m, v[u]); }
#pragma unroll
    for (int off = 16; off >= 1; off >>= 1) m = fmaxf(m, __shfl_xor(m, off, 32));
    _Float16* dst = xe + r * LDSS + lane;
#pragma unroll
    for (int u = 0; u < 8; ++u)
      dst[u * 32] = (_Float16)__builtin_amdgcn_exp2f((v[u] - m) * LOG2E);
    if (lane == 0) xmaxs[r] = m;
  }

  // ---- Phase 1b: single global pass over acc: stage f16(acc) into LDS while
  //      reducing colmax/colsum in f32; then normalize in LDS (no 2nd HBM read).
  {
    const int    j    = tid & 255;
    const int    half = tid >> 8;                         // split K range in two
    const float* ac   = A + base + (size_t)half * 128 * N256 + j;  // coalesced in j
    _Float16*    wd   = we + j * LDSS + half * 128;
    float cmax = 0.0f, csum = 0.0f;
    for (int i = 0; i < 128; ++i) {
      float a = ac[i * N256];
      cmax = fmaxf(cmax, a);
      csum += a;
      wd[i] = (_Float16)a;                                // raw stage
    }
    redmax[tid] = cmax;
    redsum[tid] = csum;
    __syncthreads();
    if (half == 0) {
      cmax = fmaxf(cmax, redmax[j + 256]);
      csum += redsum[j + 256];
      cmaxs[j] = cmax;
      cterm[j] = (__builtin_amdgcn_logf(cmax) - __builtin_amdgcn_logf(csum)) * LN2;
    }
    __syncthreads();
    const float inv = 1.0f / cmaxs[j];
    for (int i = 0; i < 128; ++i)
      wd[i] = (_Float16)((float)wd[i] * inv);             // normalize in LDS
  }
  __syncthreads();

  // ---- Phase 2: Y = E @ W, 16 waves x (64x64 tile), K=256 in steps of 32 ----
  const int wm = (wv >> 2) * 64;
  const int wn = (wv & 3) * 64;
  v8f acc[4][4];
  const v8f zero = {};
#pragma unroll
  for (int mi = 0; mi < 4; ++mi)
#pragma unroll
    for (int ni = 0; ni < 4; ++ni) acc[mi][ni] = zero;

  for (int kb = 0; kb < 256; kb += 32) {
    v16h af[4], bf[4];
#pragma unroll
    for (int mi = 0; mi < 4; ++mi) af[mi] = load_a_frag(xe, wm + mi * 16, kb, lane);
#pragma unroll
    for (int ni = 0; ni < 4; ++ni) bf[ni] = load_b_frag(we, wn + ni * 16, kb, lane);
#pragma unroll
    for (int mi = 0; mi < 4; ++mi)
#pragma unroll
      for (int ni = 0; ni < 4; ++ni)
        acc[mi][ni] = __builtin_amdgcn_wmma_f32_16x16x32_f16(
            /*neg_a=*/false, af[mi], /*neg_b=*/false, bf[ni],
            /*c_mod=*/(short)0, acc[mi][ni], /*reuse_a=*/false, /*reuse_b=*/false);
  }

  // ---- Phase 3: out = log(Y) + rowmax[b] + (log(cmax)-log(csum))[j] ----
  // Y in (~1e-3, 256]: never denormal -> raw v_log_f32 (log2) * ln2 is exact enough.
  // C layout: VGPR v, lane l -> M = v + 8*(l>=16), N = l%16.
#pragma unroll
  for (int mi = 0; mi < 4; ++mi)
#pragma unroll
    for (int ni = 0; ni < 4; ++ni) {
      v8f c = acc[mi][ni];
      const int colL = wn + ni * 16 + (lane & 15);
      const int rowB = wm + mi * 16 + ((lane >> 4) << 3);
      const float ct = cterm[colL];
#pragma unroll
      for (int v = 0; v < 8; ++v) {
        const int rowL = rowB + v;
        O[base + (size_t)rowL * N256 + colL] =
            __builtin_amdgcn_logf(c[v]) * LN2 + xmaxs[rowL] + ct;
      }
    }
}

extern "C" void kernel_launch(void* const* d_in, const int* in_sizes, int n_in,
                              void* d_out, int out_size, void* d_ws, size_t ws_size,
                              hipStream_t stream) {
  (void)in_sizes; (void)n_in; (void)d_ws; (void)ws_size; (void)out_size;
  const float* x   = (const float*)d_in[0];
  const float* acc = (const float*)d_in[1];
  float*       out = (float*)d_out;

  // 277.5 KB dynamic LDS per workgroup (limit is 320 KB/WGP on CDNA5).
  (void)hipFuncSetAttribute((const void*)densesum_logmm_wmma,
                            hipFuncAttributeMaxDynamicSharedMemorySize,
                            (int)SMEM_BYTES);

  // 32*8 = 256 independent problems, one workgroup (16 wave32) each.
  densesum_logmm_wmma<<<dim3(256), dim3(512), SMEM_BYTES, stream>>>(x, acc, out);
}
